// DeformConv_36679020707972
// MI455X (gfx1250) — matrix operand order repaired and checked by reading, not compiled, not run on hardware
//
#include <hip/hip_runtime.h>
#include <hip/hip_bf16.h>

// ---------------------------------------------------------------------------
// Deformable conv for MI455X (gfx1250, wave32, WMMA).
// Inputs (setup_inputs order):
//   d_in[0] x        (8,256,64,64) f32
//   d_in[1] offset_w (18,256,3,3)  f32
//   d_in[2] offset_b (18,)         f32
//   d_in[3] conv_w   (256,256,3,3) f32
//   d_in[4] conv_b   (256,)        f32
// Output: (8,256,64,64) f32
//
// ws layout:
//   [0, 1179648)          : conv_w as bf16, WMMA-B-fragment-swizzled
//   [2MB, 2MB + 9437184)  : per-(b,k,pixel) gather records {4 x int idx, 4 x float w}
// ---------------------------------------------------------------------------

typedef __attribute__((ext_vector_type(16))) __bf16        v16bf;
typedef __attribute__((ext_vector_type(8)))  float         v8f;
typedef __attribute__((ext_vector_type(4)))  unsigned int  v4u;
typedef __attribute__((ext_vector_type(8)))  unsigned int  v8u;

__device__ __forceinline__ unsigned short f2bf(float f) {
    unsigned u = __float_as_uint(f);
    unsigned r = u + 0x7FFFu + ((u >> 16) & 1u);   // round-to-nearest-even
    return (unsigned short)(r >> 16);
}

// ---------------------------------------------------------------------------
// Kernel 1: swizzle conv_w (O,C,3,3) f32 -> bf16 B fragments.
// Linear ws index g = ((tile*32)+lane)*16 + e, tile = (k*8+cc)*16 + nt.
// B 32x16 bf16 layout: column n = lane&15, K = (lane>>4)*16 + e.
// ---------------------------------------------------------------------------
__global__ __launch_bounds__(256) void swizzle_w_kernel(
    const float* __restrict__ cw, unsigned short* __restrict__ wsB) {
    int g    = blockIdx.x * 256 + threadIdx.x;      // < 589824
    int e    = g & 15;
    int lane = (g >> 4) & 31;
    int nt   = (g >> 9) & 15;
    int cc   = (g >> 13) & 7;
    int k    = g >> 16;                              // 0..8
    int c    = cc * 32 + ((lane >> 4) << 4) + e;     // K index (input channel)
    int o    = nt * 16 + (lane & 15);                // N index (output channel)
    float v  = cw[((size_t)o * 256 + c) * 9 + k];
    wsB[g]   = f2bf(v);
}

// ---------------------------------------------------------------------------
// Kernel 2: offset conv (256 -> 18 ch, 3x3, pad 1) + bilinear gather records.
// One block per (b, 8x8 pixel tile). x patch + offset weights staged in LDS.
// ---------------------------------------------------------------------------
__global__ __launch_bounds__(256) void offsets_kernel(
    const float* __restrict__ x, const float* __restrict__ ow,
    const float* __restrict__ ob, int* __restrict__ wsT) {
    __shared__ float patch[3200];      // [32 c][10 y][10 x]
    __shared__ float wts[5184];        // [18 ch][32 c][9 tap]
    __shared__ float offout[64 * 18];  // [pixel][ch]

    const int t    = threadIdx.x;
    const int b    = blockIdx.x >> 6;
    const int tile = blockIdx.x & 63;
    const int ty0  = (tile >> 3) * 8;
    const int tx0  = (tile & 7) * 8;
    const int p    = t & 63;           // local pixel
    const int q    = t >> 6;           // channel slot 0..3 (ch = q + 4*i)
    const int ply  = p >> 3, plx = p & 7;
    const int nch  = (q < 2) ? 5 : 4;  // #ch this thread owns (18 total)

    float acc[5] = {0.f, 0.f, 0.f, 0.f, 0.f};

    for (int cb = 0; cb < 8; ++cb) {
        const int cbase = cb * 32;
        for (int idx = t; idx < 3200; idx += 256) {
            int c = idx / 100, r = idx - c * 100;
            int yy = ty0 + r / 10 - 1;
            int xx = tx0 + (r % 10) - 1;
            float v = 0.f;
            if (yy >= 0 && yy < 64 && xx >= 0 && xx < 64)
                v = x[(((size_t)b * 256 + cbase + c) * 64 + yy) * 64 + xx];
            patch[idx] = v;
        }
        for (int idx = t; idx < 5184; idx += 256) {
            int ch = idx / 288, rem = idx - ch * 288;  // rem = c*9 + tap
            wts[idx] = ow[(size_t)ch * 2304 + (size_t)cbase * 9 + rem];
        }
        __syncthreads();
        for (int c = 0; c < 32; ++c) {
            #pragma unroll
            for (int tap = 0; tap < 9; ++tap) {
                float xv = patch[c * 100 + (ply + tap / 3) * 10 + plx + (tap % 3)];
                for (int i = 0; i < nch; ++i)
                    acc[i] += xv * wts[(q + 4 * i) * 288 + c * 9 + tap];
            }
        }
        __syncthreads();
    }

    for (int i = 0; i < nch; ++i)
        offout[p * 18 + q + 4 * i] = acc[i] + ob[q + 4 * i];
    __syncthreads();

    // Build gather records: (b, k, pixel) -> 4 indices + 4 validity-folded weights.
    for (int r = t; r < 576; r += 256) {
        int pp = r / 9, k = r - pp * 9;
        float dy = offout[pp * 18 + 2 * k];
        float dx = offout[pp * 18 + 2 * k + 1];
        int ho = ty0 + (pp >> 3), wo = tx0 + (pp & 7);
        float py = (float)(ho - 1 + k / 3) + dy;
        float px = (float)(wo - 1 + k % 3) + dx;
        float y0f = floorf(py), x0f = floorf(px);
        float wy = py - y0f, wx = px - x0f;
        int y0 = (int)y0f, x0 = (int)x0f;
        int y1 = y0 + 1, x1 = x0 + 1;
        float w00 = (1.f - wy) * (1.f - wx), w01 = (1.f - wy) * wx;
        float w10 = wy * (1.f - wx),         w11 = wy * wx;
        bool vy0 = (y0 >= 0) & (y0 < 64), vy1 = (y1 >= 0) & (y1 < 64);
        bool vx0 = (x0 >= 0) & (x0 < 64), vx1 = (x1 >= 0) & (x1 < 64);
        int cy0 = min(max(y0, 0), 63), cy1 = min(max(y1, 0), 63);
        int cx0 = min(max(x0, 0), 63), cx1 = min(max(x1, 0), 63);
        int pixg = ho * 64 + wo;
        int* dst = wsT + ((size_t)(b * 9 + k) * 4096 + pixg) * 8;
        dst[0] = cy0 * 64 + cx0;
        dst[1] = cy0 * 64 + cx1;
        dst[2] = cy1 * 64 + cx0;
        dst[3] = cy1 * 64 + cx1;
        dst[4] = __float_as_int((vy0 && vx0) ? w00 : 0.f);
        dst[5] = __float_as_int((vy0 && vx1) ? w01 : 0.f);
        dst[6] = __float_as_int((vy1 && vx0) ? w10 : 0.f);
        dst[7] = __float_as_int((vy1 && vx1) ? w11 : 0.f);
    }
}

// ---------------------------------------------------------------------------
// Kernel 3: fused sampling + WMMA GEMM, software-pipelined with a
// double-buffered bf16 A tile in LDS (one barrier per K-chunk so gathers
// for chunk i+1 overlap the 8-WMMA burst of chunk i). All 8 B fragments are
// preloaded into registers before the WMMA burst (one load clause -> MLP),
// and the next chunk's B region is prefetched into L0.
// One block per (b, 64-pixel M-tile); 8 waves:
//   wave w -> M-subtile (w&3)*16 pixels, N half (w>>2)*128 channels.
// ---------------------------------------------------------------------------
__global__ __launch_bounds__(256) void deform_main_kernel(
    const float* __restrict__ x, const unsigned short* __restrict__ wsB,
    const int* __restrict__ wsT, const float* __restrict__ cb,
    float* __restrict__ out) {
    __shared__ unsigned char smem[8192];   // 2 x 4KB A buffers / 8KB epilogue

    const int t     = threadIdx.x;
    const int lane  = t & 31;
    const int w     = t >> 5;
    const int b     = blockIdx.x >> 6;
    const int mt    = blockIdx.x & 63;
    const int p     = t >> 2;              // pixel 0..63
    const int q     = t & 3;               // 8-channel slot
    const int pix   = mt * 64 + p;
    const int msub  = w & 3;
    const int nhalf = w >> 2;

    v8f acc[8];
    #pragma unroll
    for (int i = 0; i < 8; ++i)
        #pragma unroll
        for (int j = 0; j < 8; ++j) acc[i][j] = 0.f;

    // A fragment LDS byte address (16-bit A 16x32, contiguous k order K = 16*(lane>>4) + e used here):
    // lane m<16 holds K={0..15}; lane m>=16 holds K={16..31}.
    const int amBase = (msub * 16 + (lane & 15)) * 64 + ((lane >> 4) << 5);
    const float* xb = x + (size_t)b * 256 * 4096;

    // Current gather record (per tap).
    int   i0 = 0, i1 = 0, i2 = 0, i3 = 0;
    float w0 = 0.f, w1 = 0.f, w2 = 0.f, w3 = 0.f;

    auto load_rec = [&](int k) {
        const int* rp = wsT + ((size_t)(b * 9 + k) * 4096 + pix) * 8;
        i0 = rp[0]; i1 = rp[1]; i2 = rp[2]; i3 = rp[3];
        w0 = __int_as_float(rp[4]); w1 = __int_as_float(rp[5]);
        w2 = __int_as_float(rp[6]); w3 = __int_as_float(rp[7]);
    };

    // Sample 8 channels of K-chunk kc for pixel p into A buffer (kc&1).
    auto sample = [&](int kc) {
        if ((kc & 7) == 0) load_rec(kc >> 3);
        const int cc = kc & 7;
        unsigned char* buf = smem + (kc & 1) * 4096;
        #pragma unroll
        for (int j = 0; j < 8; j += 2) {
            const int c0 = cc * 32 + q * 8 + j;
            const float* xc0 = xb + (size_t)c0 * 4096;
            const float* xc1 = xc0 + 4096;
            float v0 = w0 * xc0[i0] + w1 * xc0[i1] + w2 * xc0[i2] + w3 * xc0[i3];
            float v1 = w0 * xc1[i0] + w1 * xc1[i1] + w2 * xc1[i2] + w3 * xc1[i3];
            unsigned pk = (unsigned)f2bf(v0) | ((unsigned)f2bf(v1) << 16);
            *(unsigned*)(buf + p * 64 + (q * 8 + j) * 2) = pk;
        }
    };

    sample(0);  // prologue: fill buffer 0

    #pragma unroll 2
    for (int kc = 0; kc < 72; ++kc) {       // kc = k*8 + cc over K = 9*256
        __syncthreads();                     // A buffer (kc&1) is ready

        // Issue next chunk's gathers early; they overlap the WMMA burst below.
        if (kc + 1 < 72) sample(kc + 1);

        // Preload all 8 B fragments (pre-swizzled, 32B contiguous per lane)
        // into registers: one big load clause -> progressive loadcnt waits.
        const unsigned char* bbase =
            (const unsigned char*)wsB +
            ((size_t)((kc * 16 + nhalf * 8) * 32 + lane)) * 32;
        v8u bu[8];
        #pragma unroll
        for (int nt = 0; nt < 8; ++nt) {
            v4u blo = *(const v4u*)(bbase + nt * 1024);
            v4u bhi = *(const v4u*)(bbase + nt * 1024 + 16);
            #pragma unroll
            for (int r = 0; r < 4; ++r) { bu[nt][r] = blo[r]; bu[nt][r + 4] = bhi[r]; }
        }

        // Prefetch next chunk's B slice (16KB ahead) into L0 during the burst.
        if (kc + 1 < 72)
            __builtin_prefetch(bbase + 16384, 0, 3);

        // A fragment from LDS (two 16B loads per lane).
        const unsigned char* abuf = smem + (kc & 1) * 4096;
        v4u alo = *(const v4u*)(abuf + amBase);
        v4u ahi = *(const v4u*)(abuf + amBase + 16);
        v8u au;
        #pragma unroll
        for (int r = 0; r < 4; ++r) { au[r] = alo[r]; au[r + 4] = ahi[r]; }
        v16bf A = __builtin_bit_cast(v16bf, au);

        #pragma unroll
        for (int nt = 0; nt < 8; ++nt) {
            v16bf B = __builtin_bit_cast(v16bf, bu[nt]);
            acc[nt] = __builtin_amdgcn_wmma_f32_16x16x32_bf16(
                false, A, false, B, (short)0, acc[nt], false, false);
        }
    }

    __syncthreads();  // all A-buffer reads done before epilogue reuses LDS

    // Epilogue: LDS transpose per 16x16 tile -> coalesced 64B stores + bias.
    float* T = (float*)smem + w * 256;
    const int pix0 = mt * 64 + msub * 16;
    const int m = lane & 15;
    const int nb = lane >> 4;
    for (int nt = 0; nt < 8; ++nt) {
        #pragma unroll
        for (int r = 0; r < 8; ++r)
            T[(r + (lane >> 4) * 8) * 16 + (lane & 15)] = acc[nt][r];
        __syncthreads();
        #pragma unroll
        for (int j = 0; j < 8; ++j) {
            int n = nb + 2 * j;
            int o = nhalf * 128 + nt * 16 + n;
            float v = T[m * 16 + n] + cb[o];
            out[((size_t)b * 256 + o) * 4096 + pix0 + m] = v;
        }
        __syncthreads();
    }
}

// ---------------------------------------------------------------------------
extern "C" void kernel_launch(void* const* d_in, const int* in_sizes, int n_in,
                              void* d_out, int out_size, void* d_ws, size_t ws_size,
                              hipStream_t stream) {
    const float* x  = (const float*)d_in[0];
    const float* ow = (const float*)d_in[1];
    const float* ob = (const float*)d_in[2];
    const float* cw = (const float*)d_in[3];
    const float* cb = (const float*)d_in[4];
    float* out = (float*)d_out;

    unsigned short* wsB = (unsigned short*)d_ws;                       // 1.13 MB
    int* wsT = (int*)((unsigned char*)d_ws + (size_t)(2u << 20));      // 9.4 MB

    swizzle_w_kernel<<<2304, 256, 0, stream>>>(cw, wsB);               // 589824 elems
    offsets_kernel<<<512, 256, 0, stream>>>(x, ow, ob, wsT);           // 8 * 64 tiles
    deform_main_kernel<<<512, 256, 0, stream>>>(x, wsB, wsT, cb, out); // 8 * 64 M-tiles
}
